// MultiHeadSelfAttention_78864189489909
// MI455X (gfx1250) — compile-verified, matrix operand order checked by reading, not run on hardware
//
#include <hip/hip_runtime.h>
#include <hip/hip_bf16.h>

#define BB 4
#define NN 2048
#define HH 16
#define DD 64

typedef __attribute__((ext_vector_type(16))) __bf16 v16bf;
typedef __attribute__((ext_vector_type(8)))  float  v8f;

union FragBF { v16bf v; unsigned u[8]; uint4 q[2]; };
union Acc    { v8f v; float f[8]; };

// Pinned packed conversion: keeps SLP from turning pairs into v_perm shuffles.
__device__ __forceinline__ unsigned pk(float a, float b) {
    unsigned d;
    asm("v_cvt_pk_bf16_f32 %0, %1, %2" : "=v"(d) : "v"(a), "v"(b));
    return d;
}

__device__ __forceinline__ v8f wmma_bf16(const v16bf& a, const v16bf& b, const v8f& c) {
    return __builtin_amdgcn_wmma_f32_16x16x32_bf16(false, a, false, b, (short)0, c, false, false);
}

__device__ __forceinline__ void wait_async0() {
#if __has_builtin(__builtin_amdgcn_s_wait_asynccnt)
    __builtin_amdgcn_s_wait_asynccnt(0);
#else
    asm volatile("s_wait_asynccnt 0x0" ::: "memory");
#endif
}

// Async copy 16B (global -> LDS), GV mode, tracked by ASYNCcnt.
__device__ __forceinline__ void async_b128(unsigned lds_off, unsigned long long gaddr) {
    asm volatile("global_load_async_to_lds_b128 %0, %1, off"
                 :: "v"(lds_off), "v"(gaddr) : "memory");
}

// Fragment dword map (16-bit 16x16x32 WMMA, wave32):
//   register r (0..7), lane-half hf: dword index = (r>>2)*8 + hf*4 + (r&3)
//   within a 32-dword (64 bf16) row; K-slice ks adds 16 dwords.
// => u[0..3] and u[4..7] are each 16B-contiguous: load fragments as 2 x b128.

// ---------------------------------------------------------------------------
// Kernel 0a: convert x fp32 -> bf16 (same [B,N,H,D] layout; rows stay
// fragment-contiguous so A-fragments become pure b128 loads downstream).
// ---------------------------------------------------------------------------
__global__ __launch_bounds__(256) void pack_x_kernel(
    const float* __restrict__ x, unsigned* __restrict__ xb)
{
    const size_t i = (size_t)blockIdx.x * 256 + threadIdx.x;   // one uint4 out
    const float4* xp = (const float4*)x;
    float4 f0 = xp[i * 2 + 0];
    float4 f1 = xp[i * 2 + 1];
    uint4 o;
    o.x = pk(f0.x, f0.y); o.y = pk(f0.z, f0.w);
    o.z = pk(f1.x, f1.y); o.w = pk(f1.z, f1.w);
    ((uint4*)xb)[i] = o;
}

// ---------------------------------------------------------------------------
// Kernel 0b: pre-pack weights fp32 [H,D,D] -> bf16, fragment-major:
//   dst uint4[(h*2+ks)*256 + g*128 + hf*64 + col] = regs r=0..3 for that slot,
//   where k = ks*32 + g*16 + hf*8 + 2r (+1), col = output column.
// ---------------------------------------------------------------------------
__global__ __launch_bounds__(256) void pack_w_kernel(
    const float* __restrict__ Wq, const float* __restrict__ Wk,
    const float* __restrict__ Wv, const float* __restrict__ Wo,
    unsigned* __restrict__ Wqb, unsigned* __restrict__ Wkb,
    unsigned* __restrict__ Wvb, unsigned* __restrict__ Wob)
{
    const int idx = blockIdx.x * 256 + threadIdx.x;      // H*32*64 = 32768 total
    const int col = idx & 63;
    const int k2  = (idx >> 6) & 31;                     // k/2
    const int h   = idx >> 11;
    const int ks  = k2 >> 4;
    const int g   = (k2 >> 3) & 1;
    const int hf  = (k2 >> 2) & 1;
    const int r   = k2 & 3;
    const size_t src = ((size_t)h * DD + 2 * k2) * DD + col;
    const int dst = (((h * 2 + ks) * 256 + g * 128 + hf * 64 + col) << 2) + r;
    Wqb[dst] = pk(Wq[src], Wq[src + DD]);
    Wkb[dst] = pk(Wk[src], Wk[src + DD]);
    Wvb[dst] = pk(Wv[src], Wv[src + DD]);
    Wob[dst] = pk(Wo[src], Wo[src + DD]);
}

// One projection matrix: 16 rows x 64 cols; each B-frag = 2 x b128 loads.
// MODE 0: Q (scaled 1/32, [B,H,N,D]); 1: K ([B,H,N,D]); 2: V transposed ([B,H,D,N]).
template<int MODE>
__device__ __forceinline__ void proj_mat(
    const FragBF a[2], const uint4* __restrict__ Wb4,
    const float* __restrict__ bias, int bh, int h,
    int n0, int l16, int half, __bf16* __restrict__ dst)
{
    v8f c[4];
    { v8f z = {}; for (int i = 0; i < 4; ++i) c[i] = z; }
#pragma unroll
    for (int ks = 0; ks < 2; ++ks) {
        const uint4* wb = Wb4 + (size_t)(h * 2 + ks) * 256 + half * 64;
#pragma unroll
        for (int ct = 0; ct < 4; ++ct) {
            const int col = ct * 16 + l16;
            FragBF bf;
            bf.q[0] = wb[col];          // regs 0..3
            bf.q[1] = wb[128 + col];    // regs 4..7
            c[ct] = wmma_bf16(a[ks].v, bf.v, c[ct]);
        }
    }
#pragma unroll
    for (int ct = 0; ct < 4; ++ct) {
        const int col = ct * 16 + l16;
        const float bv = bias[h * DD + col];
#pragma unroll
        for (int r = 0; r < 8; ++r) {
            const int row = n0 + r + 8 * half;           // C: m = r + 8*half
            const float val = c[ct][r] + bv;
            if (MODE == 0)
                dst[((size_t)bh * NN + row) * DD + col] = (__bf16)(val * 0.03125f);
            else if (MODE == 1)
                dst[((size_t)bh * NN + row) * DD + col] = (__bf16)val;
            else
                dst[((size_t)bh * DD + col) * NN + row] = (__bf16)val;
        }
    }
}

// ---------------------------------------------------------------------------
// Kernel 1: fused Q/K/V projection. grid = (N/64, B*H), block = 128 (4 waves).
// ---------------------------------------------------------------------------
__global__ __launch_bounds__(128) void qkv_proj_kernel(
    const __bf16* __restrict__ xb,
    const uint4* __restrict__ Wqb, const float* __restrict__ bq,
    const uint4* __restrict__ Wkb, const float* __restrict__ bk,
    const uint4* __restrict__ Wvb, const float* __restrict__ bv,
    __bf16* __restrict__ Qs, __bf16* __restrict__ Ks, __bf16* __restrict__ Vt)
{
    const int bh   = blockIdx.y;
    const int b    = bh / HH, h = bh % HH;
    const int wave = threadIdx.x >> 5;
    const int lane = threadIdx.x & 31;
    const int half = lane >> 4;
    const int l16  = lane & 15;
    const int n0   = blockIdx.x * 64 + wave * 16;

    // A fragments: rows n0..n0+15 of xb, pure 2 x b128 per K-slice.
    FragBF a[2];
    {
        const uint4* xp4 = (const uint4*)(xb + ((size_t)(b * NN + n0 + l16) * HH + h) * DD);
#pragma unroll
        for (int ks = 0; ks < 2; ++ks) {
            a[ks].q[0] = xp4[ks * 4 + half];
            a[ks].q[1] = xp4[ks * 4 + 2 + half];
        }
    }
    proj_mat<0>(a, Wqb, bq, bh, h, n0, l16, half, Qs);
    proj_mat<1>(a, Wkb, bk, bh, h, n0, l16, half, Ks);
    proj_mat<2>(a, Wvb, bv, bh, h, n0, l16, half, Vt);
}

// Stage one 32-key K/V tile into an LDS buffer (async, ASYNCcnt-tracked).
__device__ __forceinline__ void stage_tile(
    const __bf16* __restrict__ Ks, const __bf16* __restrict__ Vt,
    int bh, int j0, int t, unsigned klds, unsigned vlds)
{
    // K tile: rows j0..j0+31 contiguous -> 4 KB, 32 B per thread
    unsigned long long kg = (unsigned long long)(uintptr_t)
        (Ks + ((size_t)bh * NN + j0) * DD) + (unsigned long long)t * 32u;
    async_b128(klds, kg);
    async_b128(klds + 16u, kg + 16u);
    // Vt tile: 64 rows (d) x 64 B; 32 B per thread
    const int vrow = t >> 1;
    unsigned long long vg = (unsigned long long)(uintptr_t)
        (Vt + ((size_t)bh * DD + vrow) * NN + j0) + ((unsigned long long)(t & 1) << 5);
    async_b128(vlds, vg);
    async_b128(vlds + 16u, vg + 16u);
}

// ---------------------------------------------------------------------------
// Kernel 2: flash attention. grid = (N/64, B*H), block = 128 (4 waves).
// Sᵀ = K·Qᵀ so the score C-layout refeeds directly as the P A-fragment.
// Double-buffered LDS staging: tile j+1 async-loads overlap tile-j compute.
// ---------------------------------------------------------------------------
__global__ __launch_bounds__(128) void flash_attn_kernel(
    const __bf16* __restrict__ Qs, const __bf16* __restrict__ Ks,
    const __bf16* __restrict__ Vt, __bf16* __restrict__ attn)
{
    __shared__ uint4 kbuf4[2][256];   // K tiles: 32 keys x 64 bf16 (4 KB each)
    __shared__ uint4 vbuf4[2][256];   // Vt tiles: 64 d x 32 bf16  (4 KB each)

    const int bh   = blockIdx.y;
    const int wave = threadIdx.x >> 5;
    const int lane = threadIdx.x & 31;
    const int half = lane >> 4;
    const int l16  = lane & 15;
    const int q0   = blockIdx.x * 64 + wave * 16;
    const int t    = threadIdx.x;

    // LDS byte offsets (generic shared pointer: low 32 bits = LDS offset)
    const unsigned koff = (unsigned)(uintptr_t)(void*)&kbuf4[0][0];
    const unsigned voff = (unsigned)(uintptr_t)(void*)&vbuf4[0][0];
    const int vrow = t >> 1;
    const unsigned kl_t = (unsigned)t * 32u;
    const unsigned vl_t = ((unsigned)vrow << 6) + ((unsigned)(t & 1) << 5);

    // Resident Qᵀ B-fragments (lane = query): 2 x b128 per K-slice.
    FragBF qt[2];
    {
        const uint4* qp4 = (const uint4*)(Qs + ((size_t)bh * NN + q0 + l16) * DD);
#pragma unroll
        for (int ks = 0; ks < 2; ++ks) {
            qt[ks].q[0] = qp4[ks * 4 + half];
            qt[ks].q[1] = qp4[ks * 4 + 2 + half];
        }
    }

    float m_run = -1e30f, l_run = 0.f;
    v8f o[4];
    { v8f z = {}; for (int i = 0; i < 4; ++i) o[i] = z; }

    // Prologue: stage tile 0 into buffer 0
    stage_tile(Ks, Vt, bh, 0, t, koff + kl_t, voff + vl_t);
    wait_async0();
    __syncthreads();

    int cur = 0;
    for (int j0 = 0; j0 < NN; j0 += 32) {
        const int nxt = cur ^ 1;
        if (j0 + 32 < NN)   // issue next tile's async loads; overlap with compute
            stage_tile(Ks, Vt, bh, j0 + 32, t,
                       koff + (unsigned)(nxt << 12) + kl_t,
                       voff + (unsigned)(nxt << 12) + vl_t);

        const uint4* kb4 = kbuf4[cur];
        const uint4* vb4 = vbuf4[cur];

        // Sᵀ tiles: st[sub] = K[j0+sub*16 .. +15] · Qᵀ   (C: lane=query, vgpr=key)
        Acc st[2];
#pragma unroll
        for (int sub = 0; sub < 2; ++sub) {
            v8f c = {};
#pragma unroll
            for (int ks = 0; ks < 2; ++ks) {
                FragBF ka;                                   // 2 x ds_load_b128
                ka.q[0] = kb4[(sub * 16 + l16) * 8 + ks * 4 + half];
                ka.q[1] = kb4[(sub * 16 + l16) * 8 + ks * 4 + 2 + half];
                c = wmma_bf16(ka.v, qt[ks].v, c);
            }
            st[sub].v = c;
        }

        // Online softmax (scale 1/32 folded into Q). Keys split lane <-> lane^16.
        float tmax = st[0].f[0];
#pragma unroll
        for (int r = 1; r < 8; ++r) tmax = fmaxf(tmax, st[0].f[r]);
#pragma unroll
        for (int r = 0; r < 8; ++r) tmax = fmaxf(tmax, st[1].f[r]);
        tmax = fmaxf(tmax, __shfl_xor(tmax, 16, 32));
        const float m_new = fmaxf(m_run, tmax);
        const float alpha = __expf(m_run - m_new);
        float tsum = 0.f;
#pragma unroll
        for (int r = 0; r < 8; ++r) { st[0].f[r] = __expf(st[0].f[r] - m_new); tsum += st[0].f[r]; }
#pragma unroll
        for (int r = 0; r < 8; ++r) { st[1].f[r] = __expf(st[1].f[r] - m_new); tsum += st[1].f[r]; }
        tsum += __shfl_xor(tsum, 16, 32);
        l_run = l_run * alpha + tsum;
        m_run = m_new;

        // Pack P as A-fragment: Sᵀ C-layout maps 1:1 onto P A-layout (same lane).
        FragBF pa;
#pragma unroll
        for (int r = 0; r < 4; ++r) {
            pa.u[r]     = pk(st[0].f[2 * r], st[0].f[2 * r + 1]);
            pa.u[4 + r] = pk(st[1].f[2 * r], st[1].f[2 * r + 1]);
        }
        // Broadcast per-query alpha into O-tile layout (query = r + 8*half)
        float am[8];
#pragma unroll
        for (int r = 0; r < 8; ++r) am[r] = __shfl(alpha, r + 8 * half, 32);

#pragma unroll
        for (int dt = 0; dt < 4; ++dt) {
            FragBF bvv;                                      // 2 x ds_load_b128
            bvv.q[0] = vb4[(dt * 16 + l16) * 4 + half];
            bvv.q[1] = vb4[(dt * 16 + l16) * 4 + 2 + half];
            v8f c = o[dt];
#pragma unroll
            for (int r = 0; r < 8; ++r) c[r] *= am[r];
            o[dt] = wmma_bf16(pa.v, bvv.v, c);
        }

        wait_async0();      // next tile's data landed in LDS
        __syncthreads();    // all waves done reading cur, next tile visible
        cur = nxt;
    }

    // Normalize by l and store attn (bf16 [B,H,N,D])
    float lm[8];
#pragma unroll
    for (int r = 0; r < 8; ++r) lm[r] = __shfl(l_run, r + 8 * half, 32);
#pragma unroll
    for (int dt = 0; dt < 4; ++dt)
#pragma unroll
        for (int r = 0; r < 8; ++r) {
            const int row = q0 + r + 8 * half;
            attn[((size_t)bh * NN + row) * DD + dt * 16 + l16] = (__bf16)(o[dt][r] / lm[r]);
        }
}

// ---------------------------------------------------------------------------
// Kernel 3: output projection. out[b,n,h,:] = attn[b,h,n,:]·Wo[h] + bo[h], fp32.
// ---------------------------------------------------------------------------
__global__ __launch_bounds__(128) void out_proj_kernel(
    const __bf16* __restrict__ attn, const uint4* __restrict__ Wob,
    const float* __restrict__ bo, float* __restrict__ out)
{
    const int bh   = blockIdx.y;
    const int b    = bh / HH, h = bh % HH;
    const int wave = threadIdx.x >> 5;
    const int lane = threadIdx.x & 31;
    const int half = lane >> 4;
    const int l16  = lane & 15;
    const int n0   = blockIdx.x * 64 + wave * 16;

    FragBF a[2];
    {
        const uint4* ap4 = (const uint4*)(attn + ((size_t)bh * NN + n0 + l16) * DD);
#pragma unroll
        for (int ks = 0; ks < 2; ++ks) {
            a[ks].q[0] = ap4[ks * 4 + half];
            a[ks].q[1] = ap4[ks * 4 + 2 + half];
        }
    }
    v8f c[4];
    { v8f z = {}; for (int i = 0; i < 4; ++i) c[i] = z; }
#pragma unroll
    for (int ks = 0; ks < 2; ++ks) {
        const uint4* wb = Wob + (size_t)(h * 2 + ks) * 256 + half * 64;
#pragma unroll
        for (int ct = 0; ct < 4; ++ct) {
            const int col = ct * 16 + l16;
            FragBF bf;
            bf.q[0] = wb[col];
            bf.q[1] = wb[128 + col];
            c[ct] = wmma_bf16(a[ks].v, bf.v, c[ct]);
        }
    }
#pragma unroll
    for (int ct = 0; ct < 4; ++ct) {
        const int col = ct * 16 + l16;
        const float bias = bo[h * DD + col];
#pragma unroll
        for (int r = 0; r < 8; ++r) {
            const int row = n0 + r + 8 * half;
            out[((size_t)(b * NN + row) * HH + h) * DD + col] = c[ct][r] + bias;
        }
    }
}

extern "C" void kernel_launch(void* const* d_in, const int* in_sizes, int n_in,
                              void* d_out, int out_size, void* d_ws, size_t ws_size,
                              hipStream_t stream) {
    const float* x  = (const float*)d_in[0];
    const float* Wq = (const float*)d_in[1];
    const float* bq = (const float*)d_in[2];
    const float* Wk = (const float*)d_in[3];
    const float* bk = (const float*)d_in[4];
    const float* Wv = (const float*)d_in[5];
    const float* bv = (const float*)d_in[6];
    const float* Wo = (const float*)d_in[7];
    const float* bo = (const float*)d_in[8];
    float* out = (float*)d_out;

    const size_t SZ = (size_t)BB * HH * NN * DD;     // elems per bf16 tensor
    const size_t WSZ = (size_t)HH * 32 * 64;         // packed-weight dwords per tensor
    __bf16* Qs = (__bf16*)d_ws;
    __bf16* Ks = Qs + SZ;
    __bf16* Vt = Ks + SZ;
    __bf16* at = Vt + SZ;
    __bf16* xb = at + SZ;
    unsigned* Wqb = (unsigned*)(xb + SZ);
    unsigned* Wkb = Wqb + WSZ;
    unsigned* Wvb = Wkb + WSZ;
    unsigned* Wob = Wvb + WSZ;

    dim3 grid(NN / 64, BB * HH);
    pack_x_kernel   <<<dim3((unsigned)(SZ / 8 / 256)), 256, 0, stream>>>(x, (unsigned*)xb);
    pack_w_kernel   <<<dim3(128), 256, 0, stream>>>(Wq, Wk, Wv, Wo, Wqb, Wkb, Wvb, Wob);
    qkv_proj_kernel <<<grid, 128, 0, stream>>>(xb, (const uint4*)Wqb, bq, (const uint4*)Wkb, bk,
                                               (const uint4*)Wvb, bv, Qs, Ks, Vt);
    flash_attn_kernel<<<grid, 128, 0, stream>>>(Qs, Ks, Vt, at);
    out_proj_kernel <<<grid, 128, 0, stream>>>(at, (const uint4*)Wob, bo, out);
}
